// CrossAttention_lucas_22514218565793
// MI455X (gfx1250) — matrix operand order repaired and checked by reading, not compiled, not run on hardware
//
#include <hip/hip_runtime.h>
#include <hip/hip_bf16.h>

typedef __attribute__((ext_vector_type(16))) __bf16 v16bf;
typedef __attribute__((ext_vector_type(8)))  float  v8f;
typedef int v4i_ __attribute__((vector_size(4 * sizeof(int))));   // matches builtin proto

#define N_SEQ  4096
#define DHEAD  64
#define NHEADS 4
#define BATCH  2
#define CIN    128
#define HID    256          // NHEADS * DHEAD
#define ATTN_SCALE 10.0f

#if defined(__has_builtin)
# if __has_builtin(__builtin_amdgcn_global_load_async_to_lds_b128)
#  define USE_ASYNC_LDS 1
# endif
#endif
#ifndef USE_ASYNC_LDS
# define USE_ASYNC_LDS 0
#endif

#define AS_GLOBAL __attribute__((address_space(1)))
#define AS_LDS    __attribute__((address_space(3)))

// ---------------------------------------------------------------------------
// Phase 1: 1x1-conv QKV projection.  raw[b][768][n] = W_qkv[oc] . src[b][:][n]
// oc in [0,256): q from x ; [256,512): k from y ; [512,768): v from y
// ---------------------------------------------------------------------------
__global__ __launch_bounds__(256) void k_qkv_proj(const float* __restrict__ x,
                                                  const float* __restrict__ y,
                                                  const float* __restrict__ Wqkv,
                                                  float* __restrict__ raw) {
  const int n  = blockIdx.x * 256 + threadIdx.x;   // 0..4095 (coalesced)
  const int oc = blockIdx.y;                        // 0..767 (uniform -> scalar W loads)
  const int b  = blockIdx.z;                        // 0..1
  const float* src = (oc < HID) ? x : y;
  const float* X = src + (size_t)b * CIN * N_SEQ;
  const float* w = Wqkv + (size_t)oc * CIN;
  float acc = 0.f;
  #pragma unroll 8
  for (int c = 0; c < CIN; ++c) acc = fmaf(w[c], X[(size_t)c * N_SEQ + n], acc);
  raw[((size_t)b * (3 * HID) + oc) * N_SEQ + n] = acc;
}

// ---------------------------------------------------------------------------
// Phase 2: per-row L2 norm along spatial axis (q,k only) + bf16 convert into
// attention layouts: Q,K -> [bh][n][d] (d contiguous), V -> [bh][d][n].
// One workgroup per (b, channel) row of 4096 floats.
// ---------------------------------------------------------------------------
__global__ __launch_bounds__(256) void k_norm_cvt(const float* __restrict__ raw,
                                                  __bf16* __restrict__ Qbf,
                                                  __bf16* __restrict__ Kbf,
                                                  __bf16* __restrict__ Vtb) {
  const int row  = blockIdx.x;            // 0 .. 2*768-1
  const int b    = row / (3 * HID);
  const int ch   = row % (3 * HID);
  const int kind = ch / HID;              // 0=q 1=k 2=v
  const int c    = ch % HID;
  const int h    = c / DHEAD, d = c % DHEAD;
  const int bh   = b * NHEADS + h;
  const float* src = raw + (size_t)row * N_SEQ;

  __shared__ float red[256];
  float scale = 1.0f;
  if (kind < 2) {
    float ss = 0.f;
    for (int n = threadIdx.x; n < N_SEQ; n += 256) { float v = src[n]; ss = fmaf(v, v, ss); }
    red[threadIdx.x] = ss;
    __syncthreads();
    #pragma unroll
    for (int s = 128; s > 0; s >>= 1) {
      if ((int)threadIdx.x < s) red[threadIdx.x] += red[threadIdx.x + s];
      __syncthreads();
    }
    const float norm = sqrtf(red[0]);
    scale = 1.0f / fmaxf(norm, 1e-12f);
  }

  if (kind == 0) {
    for (int n = threadIdx.x; n < N_SEQ; n += 256)
      Qbf[((size_t)bh * N_SEQ + n) * DHEAD + d] = (__bf16)(src[n] * scale);
  } else if (kind == 1) {
    for (int n = threadIdx.x; n < N_SEQ; n += 256)
      Kbf[((size_t)bh * N_SEQ + n) * DHEAD + d] = (__bf16)(src[n] * scale);
  } else {
    for (int n = threadIdx.x; n < N_SEQ; n += 256)
      Vtb[((size_t)bh * DHEAD + d) * N_SEQ + n] = (__bf16)src[n];
  }
}

// ---------------------------------------------------------------------------
// Tile fill: K tile [64j][64d] (contiguous 8KB) and V^T tile [64d][64j]
// (64 rows of 128B). Async memory->LDS (ASYNCcnt) when available, else
// VGPR-staged copies. 128 threads x 4 x b128 per tensor.
// ---------------------------------------------------------------------------
__device__ __forceinline__ void tile_load(const __bf16* __restrict__ Kb,
                                          const __bf16* __restrict__ Vb,
                                          int j0,
                                          __bf16 (*Klb)[64], __bf16 (*Vlb)[64],
                                          int tid) {
#if USE_ASYNC_LDS
  const uint4* gk = (const uint4*)(Kb + (size_t)j0 * DHEAD);
  for (int t = tid; t < 512; t += 128) {
    __builtin_amdgcn_global_load_async_to_lds_b128(
        (AS_GLOBAL v4i_*)(gk + t),
        (AS_LDS v4i_*)((uint4*)(&Klb[0][0]) + t), 0, 0);
  }
  for (int t = tid; t < 512; t += 128) {
    const int d = t >> 3, seg = t & 7;
    __builtin_amdgcn_global_load_async_to_lds_b128(
        (AS_GLOBAL v4i_*)(Vb + (size_t)d * N_SEQ + j0 + seg * 8),
        (AS_LDS v4i_*)(&Vlb[d][seg * 8]), 0, 0);
  }
#else
  const uint4* gk = (const uint4*)(Kb + (size_t)j0 * DHEAD);
  uint4* lk = (uint4*)(&Klb[0][0]);
  for (int t = tid; t < 512; t += 128) lk[t] = gk[t];
  for (int t = tid; t < 512; t += 128) {
    const int d = t >> 3, seg = t & 7;
    ((uint4*)(&Vlb[d][0]))[seg] = *(const uint4*)(Vb + (size_t)d * N_SEQ + j0 + seg * 8);
  }
#endif
}

__device__ __forceinline__ void tile_wait() {
#if USE_ASYNC_LDS
# if defined(__has_builtin) && __has_builtin(__builtin_amdgcn_s_wait_asynccnt)
  __builtin_amdgcn_s_wait_asynccnt(0);
# else
  asm volatile("s_wait_asynccnt 0" ::: "memory");
# endif
#endif
}

// ---------------------------------------------------------------------------
// Phase 3: fused flash attention, WMMA bf16, double-buffered async LDS tiles.
// Grid: (N/64 i-tiles, B*H). Block: 128 threads = 4 wave32; each wave owns a
// 16-row i-subtile. j streamed in 64-wide tiles.
// ---------------------------------------------------------------------------
__global__ __launch_bounds__(128) void k_flash_attn(const __bf16* __restrict__ Q,
                                                    const __bf16* __restrict__ K,
                                                    const __bf16* __restrict__ Vtb,
                                                    float* __restrict__ O) {
  const int bh   = blockIdx.y;
  const int i0   = blockIdx.x * 64;
  const int wv   = threadIdx.x >> 5;     // wave id 0..3
  const int lane = threadIdx.x & 31;
  const int hlf  = lane >> 4;            // 0 or 1 (lane half)
  const int l16  = lane & 15;

  const __bf16* Qb = Q   + (size_t)bh * N_SEQ * DHEAD;
  const __bf16* Kb = K   + (size_t)bh * N_SEQ * DHEAD;
  const __bf16* Vb = Vtb + (size_t)bh * DHEAD * N_SEQ;

  __shared__ __attribute__((aligned(32))) __bf16 Kl[2][64][64];   // [buf][j][d]
  __shared__ __attribute__((aligned(32))) __bf16 Vl[2][64][64];   // [buf][d][j]
  __shared__ __attribute__((aligned(32))) __bf16 Pl[4][16][64];   // per-wave P [i][j]

  // Persistent Q A-fragments: 16 rows x 64 d -> two 16x32 bf16 frags.
  // A layout: lane L row M=L%16 holds K in {base..base+15},
  // base = 16*(L/16).
  v16bf aq[2];
  {
    const __bf16* qrow = Qb + (size_t)(i0 + wv * 16 + l16) * DHEAD;
    #pragma unroll
    for (int c = 0; c < 2; ++c) {
      const int base = c * 32 + hlf * 16;
      #pragma unroll
      for (int t = 0; t < 8; ++t) {
        aq[c][t]     = qrow[base + t];
        aq[c][8 + t] = qrow[base + 8 + t];
      }
    }
  }

  v8f o_acc[4];                 // O accumulator, C layout, 4 x (16i x 16d)
  #pragma unroll
  for (int c = 0; c < 4; ++c)
    #pragma unroll
    for (int t = 0; t < 8; ++t) o_acc[c][t] = 0.f;
  float m_r[8], l_r[8];         // per-row running max / sum (row = r + 8*hlf)
  #pragma unroll
  for (int r = 0; r < 8; ++r) { m_r[r] = -1e30f; l_r[r] = 0.f; }

  // prologue: start tile 0 into buffer 0
  tile_load(Kb, Vb, 0, Kl[0], Vl[0], threadIdx.x);

  const int NT = N_SEQ / 64;
  for (int jt = 0; jt < NT; ++jt) {
    const int cur = jt & 1;

    // tile jt's async fills done (per-wave), then make LDS visible block-wide
    tile_wait();
    __syncthreads();

    // kick off tile jt+1 into the other buffer (safe: the barrier above proves
    // every wave finished reading that buffer in iteration jt-1)
    if (jt + 1 < NT)
      tile_load(Kb, Vb, (jt + 1) * 64, Kl[cur ^ 1], Vl[cur ^ 1], threadIdx.x);
    if (jt + 2 < NT)   // warm L2 two tiles ahead
      __builtin_prefetch(Kb + (size_t)(jt + 2) * 64 * DHEAD + (size_t)threadIdx.x * 32, 0, 0);

    __bf16 (*Klb)[64] = Kl[cur];
    __bf16 (*Vlb)[64] = Vl[cur];

    // --- S = Q . K^T : 4 j-subtiles, 2 WMMAs each (d split 2x32) ---
    v8f s[4];
    #pragma unroll
    for (int jj = 0; jj < 4; ++jj) {
      v8f acc = {0.f, 0.f, 0.f, 0.f, 0.f, 0.f, 0.f, 0.f};
      #pragma unroll
      for (int c = 0; c < 2; ++c) {
        // B frag (32d x 16j): lane = column j, 16 contiguous d per lane half
        const v16bf bk = *(const v16bf*)(&Klb[jj * 16 + l16][c * 32 + hlf * 16]);
        acc = __builtin_amdgcn_wmma_f32_16x16x32_bf16(false, aq[c], false, bk,
                                                      (short)0, acc, false, false);
      }
      s[jj] = acc;
    }

    // --- online softmax: row max over 64 j (4 subtiles x 16-lane group) ---
    float alpha[8];
    #pragma unroll
    for (int r = 0; r < 8; ++r) {
      float lm = -1e30f;
      #pragma unroll
      for (int jj = 0; jj < 4; ++jj) lm = fmaxf(lm, s[jj][r]);
      lm *= ATTN_SCALE;
      #pragma unroll
      for (int off = 8; off >= 1; off >>= 1)
        lm = fmaxf(lm, __shfl_xor(lm, off, 32));
      const float m_new = fmaxf(m_r[r], lm);
      alpha[r] = __expf(m_r[r] - m_new);
      m_r[r] = m_new;
    }

    // --- P = exp(S*scale - m): write per-wave LDS tile + row sums ---
    #pragma unroll
    for (int r = 0; r < 8; ++r) {
      float ls = 0.f;
      #pragma unroll
      for (int jj = 0; jj < 4; ++jj) {
        const float p = __expf(s[jj][r] * ATTN_SCALE - m_r[r]);
        ls += p;
        Pl[wv][r + 8 * hlf][jj * 16 + l16] = (__bf16)p;   // C layout -> [i][j]
      }
      #pragma unroll
      for (int off = 8; off >= 1; off >>= 1)
        ls += __shfl_xor(ls, off, 32);
      l_r[r] = l_r[r] * alpha[r] + ls;
    }

    // rescale O accumulators by alpha
    #pragma unroll
    for (int c = 0; c < 4; ++c)
      #pragma unroll
      for (int r = 0; r < 8; ++r) o_acc[c][r] *= alpha[r];

    // this wave's P stores must land before its own ds reads below
    asm volatile("s_wait_dscnt 0" ::: "memory");

    // --- O += P . V : A = P frags (16i x 32j), B = V^T frags from Vl ---
    v16bf ap[2];
    #pragma unroll
    for (int jc = 0; jc < 2; ++jc) {
      const __bf16* prow = &Pl[wv][l16][0];
      const int base = jc * 32 + hlf * 16;
      #pragma unroll
      for (int t = 0; t < 8; ++t) {
        ap[jc][t]     = prow[base + t];
        ap[jc][8 + t] = prow[base + 8 + t];
      }
    }
    #pragma unroll
    for (int c = 0; c < 4; ++c) {
      #pragma unroll
      for (int jc = 0; jc < 2; ++jc) {
        // B frag (32j x 16d): lane = column d, 16 contiguous j per lane half
        const v16bf bv = *(const v16bf*)(&Vlb[c * 16 + l16][jc * 32 + hlf * 16]);
        o_acc[c] = __builtin_amdgcn_wmma_f32_16x16x32_bf16(false, ap[jc], false, bv,
                                                           (short)0, o_acc[c], false, false);
      }
    }
  }

  // --- epilogue: O /= l, store [bh][i][d] fp32 ---
  float* Ob = O + (size_t)bh * N_SEQ * DHEAD;
  #pragma unroll
  for (int c = 0; c < 4; ++c) {
    #pragma unroll
    for (int r = 0; r < 8; ++r) {
      const int row = i0 + wv * 16 + r + 8 * hlf;
      const int d   = c * 16 + l16;
      Ob[(size_t)row * DHEAD + d] = o_acc[c][r] / l_r[r];
    }
  }
}

// ---------------------------------------------------------------------------
// Phase 4: output 1x1 conv: out[b][o][n] = W_out[o] . O[b][:][n] + b_out[o]
// ---------------------------------------------------------------------------
__global__ __launch_bounds__(256) void k_out_proj(const float* __restrict__ Og,
                                                  const float* __restrict__ Wout,
                                                  const float* __restrict__ bout,
                                                  float* __restrict__ out) {
  const int n = blockIdx.x * 256 + threadIdx.x;
  const int o = blockIdx.y;        // 0..127 (uniform)
  const int b = blockIdx.z;
  float acc = bout[o];
  #pragma unroll
  for (int h = 0; h < NHEADS; ++h) {
    const float* Ob = Og + ((size_t)(b * NHEADS + h) * N_SEQ + n) * DHEAD;
    const float* w  = Wout + (size_t)o * HID + h * DHEAD;
    #pragma unroll 8
    for (int d = 0; d < DHEAD; ++d) acc = fmaf(w[d], Ob[d], acc);
  }
  out[((size_t)b * CIN + o) * N_SEQ + n] = acc;
}

// ---------------------------------------------------------------------------
extern "C" void kernel_launch(void* const* d_in, const int* in_sizes, int n_in,
                              void* d_out, int out_size, void* d_ws, size_t ws_size,
                              hipStream_t stream) {
  const float* x    = (const float*)d_in[0];
  const float* y    = (const float*)d_in[1];
  const float* Wqkv = (const float*)d_in[2];
  const float* Wout = (const float*)d_in[3];
  const float* bout = (const float*)d_in[4];
  float* out = (float*)d_out;

  char* ws = (char*)d_ws;
  const size_t raw_bytes = (size_t)BATCH * 3 * HID * N_SEQ * sizeof(float);            // 24 MB
  const size_t bf_bytes  = (size_t)BATCH * NHEADS * N_SEQ * DHEAD * sizeof(__bf16);    // 4 MB
  float*  raw = (float*)ws;
  __bf16* Qbf = (__bf16*)(ws + raw_bytes);
  __bf16* Kbf = (__bf16*)(ws + raw_bytes + bf_bytes);
  __bf16* Vtb = (__bf16*)(ws + raw_bytes + 2 * bf_bytes);
  float*  Og  = (float*) (ws + raw_bytes + 3 * bf_bytes);                              // 8 MB

  k_qkv_proj  <<<dim3(N_SEQ / 256, 3 * HID, BATCH), 256, 0, stream>>>(x, y, Wqkv, raw);
  k_norm_cvt  <<<dim3(BATCH * 3 * HID),             256, 0, stream>>>(raw, Qbf, Kbf, Vtb);
  k_flash_attn<<<dim3(N_SEQ / 64, BATCH * NHEADS),  128, 0, stream>>>(Qbf, Kbf, Vtb, Og);
  k_out_proj  <<<dim3(N_SEQ / 256, CIN, BATCH),     256, 0, stream>>>(Og, Wout, bout, out);
}
